// CausalSelfAttention_17626545782991
// MI455X (gfx1250) — matrix operand order repaired and checked by reading, not compiled, not run on hardware
//
#include <hip/hip_runtime.h>

// CausalSelfAttention forward for MI455X (gfx1250, wave32, WMMA).
// B=4, S=2048, H=1024, NH=16, HD=64.
#define BB 4
#define SS 2048
#define HH 1024
#define NHD 16
#define HD 64
#define MT (BB * SS) // 8192 rows of X

typedef __bf16 bf16;
typedef __attribute__((ext_vector_type(16))) __bf16 v16bf;
typedef __attribute__((ext_vector_type(8)))  __bf16 v8bf;
typedef __attribute__((ext_vector_type(8)))  float  v8f;

union Frag { v16bf v; v8bf h[2]; };

static __device__ inline v8f vzero8() {
  v8f z = {0.f, 0.f, 0.f, 0.f, 0.f, 0.f, 0.f, 0.f};
  return z;
}

static __device__ inline v8f wmma_bf16(v16bf a, v16bf b, v8f c) {
  // (neg_a, A, neg_b, B, c_mod, C, reuse_a, reuse_b)
  return __builtin_amdgcn_wmma_f32_16x16x32_bf16(false, a, false, b, (short)0, c,
                                                 false, false);
}

#define NEG_BIG (-3.0e38f)

// ---------------------------------------------------------------------------
// f32 -> bf16 conversion (bandwidth trivial vs 23.3 TB/s; runs once)
// ---------------------------------------------------------------------------
__global__ void cvt_bf16_kernel(const float* __restrict__ src,
                                bf16* __restrict__ dst, int n) {
  int i = (blockIdx.x * blockDim.x + threadIdx.x) * 4;
  const int stride = gridDim.x * blockDim.x * 4;
  for (; i + 3 < n; i += stride) {
    float4 f = *(const float4*)(src + i);
    dst[i + 0] = (bf16)f.x;
    dst[i + 1] = (bf16)f.y;
    dst[i + 2] = (bf16)f.z;
    dst[i + 3] = (bf16)f.w;
  }
}

// ---------------------------------------------------------------------------
// QKV projection: out = X @ W^T + b, via v_wmma_f32_16x16x32_bf16.
// Block = 8 waves; wave w computes rows [m0, m0+16) x cols [n0, n0+64).
// blockIdx.z selects q/k/v. Q,K stored [B,NH,S,HD]; V stored transposed
// [B,NH,HD,S] so the PV-matmul B-fragments load contiguously.
// ---------------------------------------------------------------------------
__global__ __launch_bounds__(256) void qkv_gemm_kernel(
    const bf16* __restrict__ X,
    const bf16* __restrict__ Wq, const bf16* __restrict__ Wk,
    const bf16* __restrict__ Wv,
    const float* __restrict__ bq, const float* __restrict__ bk,
    const float* __restrict__ bv,
    bf16* __restrict__ Qh, bf16* __restrict__ Kh, bf16* __restrict__ Vt) {
  const int nt  = blockIdx.x;  // 0..15  (N tiles of 64)
  const int mtl = blockIdx.y;  // 0..63  (M tiles of 128)
  const int mat = blockIdx.z;  // 0=q 1=k 2=v
  const bf16*  W    = (mat == 0) ? Wq : (mat == 1) ? Wk : Wv;
  const float* bias = (mat == 0) ? bq : (mat == 1) ? bk : bv;

  const int wave = threadIdx.x >> 5;
  const int lane = threadIdx.x & 31;
  const int l15  = lane & 15;
  const int lh   = lane >> 4;

  const int m0 = mtl * 128 + wave * 16;
  const int n0 = nt * 64;

  // A fragment (16x32 bf16): lane -> row l15; chunks at k+(lh*8) and +16.
  const bf16* aptr = X + (size_t)(m0 + l15) * HH + lh * 8;
  // B fragment (32x16 bf16): lane -> col l15; chunks at k+(lh*8) and +16.
  // B = W^T, so column n of B = row n of W (contiguous in k). 
  const bf16* bptr[4];
#pragma unroll
  for (int j = 0; j < 4; ++j)
    bptr[j] = W + (size_t)(n0 + j * 16 + l15) * HH + lh * 8;

  v8f acc[4];
#pragma unroll
  for (int j = 0; j < 4; ++j) acc[j] = vzero8();

  for (int k = 0; k < HH; k += 32) {
    Frag a;
    a.h[0] = *(const v8bf*)(aptr + k);
    a.h[1] = *(const v8bf*)(aptr + k + 16);
#pragma unroll
    for (int j = 0; j < 4; ++j) {
      Frag b;
      b.h[0] = *(const v8bf*)(bptr[j] + k);
      b.h[1] = *(const v8bf*)(bptr[j] + k + 16);
      acc[j] = wmma_bf16(a.v, b.v, acc[j]);
    }
  }

  // Epilogue: bias add, split heads, store bf16.
#pragma unroll
  for (int j = 0; j < 4; ++j) {
    const int n  = n0 + j * 16 + l15;
    const int hh = n >> 6;
    const int d  = n & 63;
    const float bb = bias[n];
#pragma unroll
    for (int v = 0; v < 8; ++v) {
      const int m    = m0 + v + lh * 8; // C layout: row = v + 8*(lane>=16)
      const int bidx = m >> 11;
      const int s    = m & (SS - 1);
      const float val = acc[j][v] + bb;
      if (mat == 2) {
        Vt[((size_t)(bidx * NHD + hh) * HD + d) * SS + s] = (bf16)val;
      } else {
        bf16* dst = (mat == 0) ? Qh : Kh;
        dst[((size_t)(bidx * NHD + hh) * SS + s) * HD + d] = (bf16)val;
      }
    }
  }
}

// ---------------------------------------------------------------------------
// Flash-style causal attention. One wave per 16-query tile; streams 64-key
// blocks with online softmax (fp32), WMMA for QK^T and PV (bf16 x bf16 -> f32).
// P goes through a per-wave 2KB LDS tile to re-layout C->A fragments.
// ---------------------------------------------------------------------------
__global__ __launch_bounds__(256) void attn_kernel(
    const bf16* __restrict__ Qh, const bf16* __restrict__ Kh,
    const bf16* __restrict__ Vt, const float* __restrict__ amask,
    float* __restrict__ out) {
  __shared__ bf16 plds[8][16 * 64];

  const int wave = threadIdx.x >> 5;
  const int lane = threadIdx.x & 31;
  const int l15  = lane & 15;
  const int lh   = lane >> 4;

  const int tile = blockIdx.x * 8 + wave; // 0..8191
  const int qt = tile & (SS / 16 - 1);    // query tile in sequence
  const int bh = tile >> 7;               // (b, head)
  const int b  = bh >> 4;
  const int hh = bh & 15;
  const int q0 = qt * 16;

  const bf16* Qp = Qh + (size_t)bh * SS * HD;
  const bf16* Kp = Kh + (size_t)bh * SS * HD;
  const bf16* Vp = Vt + (size_t)bh * HD * SS;
  const float* mp = amask + b * SS;

  // Q A-fragments for d=[0,32) and d=[32,64): held in registers all loop.
  Frag qa[2];
#pragma unroll
  for (int ks = 0; ks < 2; ++ks) {
    const bf16* p = Qp + (size_t)(q0 + l15) * HD + ks * 32 + lh * 8;
    qa[ks].h[0] = *(const v8bf*)p;
    qa[ks].h[1] = *(const v8bf*)(p + 16);
  }

  v8f o[4];
  float mrun[8], lrun[8];
#pragma unroll
  for (int od = 0; od < 4; ++od) o[od] = vzero8();
#pragma unroll
  for (int v = 0; v < 8; ++v) { mrun[v] = NEG_BIG; lrun[v] = 0.f; }

  bf16* pl = &plds[wave][0];

  for (int kb = 0; kb <= q0 + 15; kb += 64) {
    // --- scores: S = Q K^T (16 x 64), 8 WMMAs ---
    v8f sc[4];
#pragma unroll
    for (int jf = 0; jf < 4; ++jf) sc[jf] = vzero8();
#pragma unroll
    for (int jf = 0; jf < 4; ++jf) {
      const bf16* kp = Kp + (size_t)(kb + jf * 16 + l15) * HD + lh * 8;
#pragma unroll
      for (int ks = 0; ks < 2; ++ks) {
        Frag kf;
        kf.h[0] = *(const v8bf*)(kp + ks * 32);
        kf.h[1] = *(const v8bf*)(kp + ks * 32 + 16);
        sc[jf] = wmma_bf16(qa[ks].v, kf.v, sc[jf]);
      }
    }

    // --- mask + additive attention_mask + 1/sqrt(HD), in place ---
    const bool bnd = (kb + 63 > q0); // boundary block needs causal mask
#pragma unroll
    for (int jf = 0; jf < 4; ++jf) {
      const int j = kb + jf * 16 + l15;
      const float mk = mp[j];
#pragma unroll
      for (int v = 0; v < 8; ++v) {
        float z = (sc[jf][v] + mk) * 0.125f; // /sqrt(64)
        if (bnd && (j > q0 + v + lh * 8)) z = NEG_BIG;
        sc[jf][v] = z;
      }
    }

    // --- online softmax: row max (half-wave shuffle reduce) ---
    float pm[8];
#pragma unroll
    for (int v = 0; v < 8; ++v)
      pm[v] = fmaxf(fmaxf(sc[0][v], sc[1][v]), fmaxf(sc[2][v], sc[3][v]));
#pragma unroll
    for (int msk = 1; msk <= 8; msk <<= 1)
#pragma unroll
      for (int v = 0; v < 8; ++v)
        pm[v] = fmaxf(pm[v], __shfl_xor(pm[v], msk, 32));

    float sf[8];
#pragma unroll
    for (int v = 0; v < 8; ++v) {
      const float mn = fmaxf(mrun[v], pm[v]);
      sf[v] = __expf(mrun[v] - mn);
      mrun[v] = mn;
    }

    // --- P = exp(z - m), row sums, P -> bf16 LDS tile (16x64, row-major) ---
    float rs[8];
#pragma unroll
    for (int v = 0; v < 8; ++v) rs[v] = 0.f;
#pragma unroll
    for (int jf = 0; jf < 4; ++jf)
#pragma unroll
      for (int v = 0; v < 8; ++v) {
        const float p = __expf(sc[jf][v] - mrun[v]);
        rs[v] += p;
        pl[(v + lh * 8) * 64 + jf * 16 + l15] = (bf16)p;
      }
#pragma unroll
    for (int msk = 1; msk <= 8; msk <<= 1)
#pragma unroll
      for (int v = 0; v < 8; ++v) rs[v] += __shfl_xor(rs[v], msk, 32);
#pragma unroll
    for (int v = 0; v < 8; ++v) lrun[v] = lrun[v] * sf[v] + rs[v];
#pragma unroll
    for (int od = 0; od < 4; ++od)
#pragma unroll
      for (int v = 0; v < 8; ++v) o[od][v] *= sf[v];

    // --- O += P V (16 x 64 keys x 64 d), 8 WMMAs; P re-read as A-frags ---
#pragma unroll
    for (int ks = 0; ks < 2; ++ks) {
      Frag pa;
      const bf16* lp = pl + l15 * 64 + ks * 32 + lh * 8;
      pa.h[0] = *(const v8bf*)lp;
      pa.h[1] = *(const v8bf*)(lp + 16);
#pragma unroll
      for (int od = 0; od < 4; ++od) {
        // V^T layout: column d of B is contiguous along keys.
        const bf16* vp = Vp + (size_t)(od * 16 + l15) * SS + kb + ks * 32 + lh * 8;
        Frag vb;
        vb.h[0] = *(const v8bf*)vp;
        vb.h[1] = *(const v8bf*)(vp + 16);
        o[od] = wmma_bf16(pa.v, vb.v, o[od]);
      }
    }
  }

  // --- epilogue: merge heads, divide by l, store f32 ---
#pragma unroll
  for (int od = 0; od < 4; ++od)
#pragma unroll
    for (int v = 0; v < 8; ++v) {
      const int s = q0 + v + lh * 8;
      out[(size_t)(b * SS + s) * HH + hh * 64 + od * 16 + l15] =
          o[od][v] / lrun[v];
    }
}

// ---------------------------------------------------------------------------
extern "C" void kernel_launch(void* const* d_in, const int* in_sizes, int n_in,
                              void* d_out, int out_size, void* d_ws,
                              size_t ws_size, hipStream_t stream) {
  const float* hs    = (const float*)d_in[0];
  const float* amask = (const float*)d_in[1];
  const float* Wq    = (const float*)d_in[2];
  const float* bq    = (const float*)d_in[3];
  const float* Wk    = (const float*)d_in[4];
  const float* bk    = (const float*)d_in[5];
  const float* Wv    = (const float*)d_in[6];
  const float* bv    = (const float*)d_in[7];
  float* out = (float*)d_out;

  // Workspace layout (bf16 elements): X | Wq | Wk | Wv | Q | K | Vt  (~70 MB)
  bf16* Xb  = (bf16*)d_ws;
  bf16* Wqb = Xb + (size_t)MT * HH;
  bf16* Wkb = Wqb + (size_t)HH * HH;
  bf16* Wvb = Wkb + (size_t)HH * HH;
  bf16* Qh  = Wvb + (size_t)HH * HH;
  bf16* Kh  = Qh + (size_t)MT * HH;
  bf16* Vt  = Kh + (size_t)MT * HH;

  cvt_bf16_kernel<<<2048, 256, 0, stream>>>(hs, Xb, MT * HH);
  cvt_bf16_kernel<<<512, 256, 0, stream>>>(Wq, Wqb, HH * HH);
  cvt_bf16_kernel<<<512, 256, 0, stream>>>(Wk, Wkb, HH * HH);
  cvt_bf16_kernel<<<512, 256, 0, stream>>>(Wv, Wvb, HH * HH);

  qkv_gemm_kernel<<<dim3(16, 64, 3), 256, 0, stream>>>(
      Xb, Wqb, Wkb, Wvb, bq, bk, bv, Qh, Kh, Vt);

  attn_kernel<<<1024, 256, 0, stream>>>(Qh, Kh, Vt, amask, out);
}
